// DynamicAttention_multiview_11630771437667
// MI455X (gfx1250) — matrix operand order repaired and checked by reading, not compiled, not run on hardware
//
#include <hip/hip_runtime.h>
#include <math.h>

// Swin-V2 style dual-view window cross-attention for MI455X (gfx1250).
// B=512 windows, N=64 tokens, DIM=256, NH=8 heads, HD=32.
// All GEMMs via v_wmma_f32_16x16x32_bf16 (wave32, 16x16 tiles, K=32).
// Key staging in the attention kernel uses CDNA5 async memory->LDS copies
// (global_load_async_to_lds_b128, drained with s_wait_asynccnt).

typedef __attribute__((ext_vector_type(16))) __bf16 bf16x16;
typedef __attribute__((ext_vector_type(8)))  __bf16 bf16x8;
typedef __attribute__((ext_vector_type(8)))  float  f32x8;

static __device__ __forceinline__ f32x8 wmma_bf16(bf16x16 a, bf16x16 b, f32x8 c) {
  return __builtin_amdgcn_wmma_f32_16x16x32_bf16(false, a, false, b, (short)0, c,
                                                 false, false);
}

// A-matrix 16x32 bf16 fragment (M rows, K cols) from row-major [.,stride] source.
// Layout used here (contiguous k, same as B): lane half h, element e -> K = h*16 + e, M = lane&15.
static __device__ __forceinline__ bf16x16 load_a_frag(const __bf16* base, int stride,
                                                      int mbase, int kbase, int lane) {
  const int half = lane >> 4;
  const int m = lane & 15;
  const __bf16* p = base + (size_t)(mbase + m) * stride + kbase + half * 16;
  bf16x8 lo = *(const bf16x8*)(p);
  bf16x8 hi = *(const bf16x8*)(p + 8);
  bf16x16 a;
#pragma unroll
  for (int i = 0; i < 8; ++i) { a[i] = lo[i]; a[8 + i] = hi[i]; }
  return a;
}

// B-matrix 32x16 bf16 fragment (K rows, N cols) where memory holds B transposed
// row-major (rows = N, cols = K). Layout used here (contiguous k, matches A): K = half*16 + e, N = lane&15.
static __device__ __forceinline__ bf16x16 load_b_frag(const __bf16* base, int stride,
                                                      int nbase, int kbase, int lane) {
  const int half = lane >> 4;
  const int n = lane & 15;
  return *(const bf16x16*)(base + (size_t)(nbase + n) * stride + kbase + half * 16);
}

// CDNA5 async memory->LDS copy, 16 bytes per active lane (ASYNCcnt tracked).
// VDST VGPR = per-lane LDS byte address,
// VADDR = 64-bit global byte address.
static __device__ __forceinline__ void async_copy_b128(unsigned lds_off,
                                                       const void* gptr) {
  asm volatile("global_load_async_to_lds_b128 %0, %1, off"
               :
               : "v"(lds_off), "v"((unsigned long long)(uintptr_t)gptr)
               : "memory");
}

static __device__ __forceinline__ void wait_asynccnt0() {
  asm volatile("s_wait_asynccnt 0x0" ::: "memory");
}

// ---------------------------------------------------------------------------
// Kernel 0: weight bf16 casts + CPB relative-position bias (16*sigmoid) table.
// ---------------------------------------------------------------------------
__global__ __launch_bounds__(256) void prep_kernel(
    const float* __restrict__ qkv_w, const float* __restrict__ proj_w,
    const float* __restrict__ cw1, const float* __restrict__ cb1,
    const float* __restrict__ cw2,
    __bf16* __restrict__ qkv_bf, __bf16* __restrict__ proj_bf,
    float* __restrict__ bias16) {
  __shared__ float tbl[225][8];
  const int t = threadIdx.x;

  for (int i = t; i < 768 * 256; i += 256) qkv_bf[i] = (__bf16)qkv_w[i];
  for (int i = t; i < 256 * 256; i += 256) proj_bf[i] = (__bf16)proj_w[i];

  // Continuous position bias MLP: relu(table @ cw1^T + cb1) @ cw2^T
  for (int p = t; p < 225; p += 256) {
    const int i = p / 15, j = p % 15;
    float c0 = 8.0f * (float)(i - 7) / 7.0f;
    float c1 = 8.0f * (float)(j - 7) / 7.0f;
    float s0 = (c0 > 0.f) ? 1.f : ((c0 < 0.f) ? -1.f : 0.f);
    float s1 = (c1 > 0.f) ? 1.f : ((c1 < 0.f) ? -1.f : 0.f);
    float v0 = s0 * log2f(fabsf(c0) + 1.0f) * (1.0f / 3.0f);
    float v1 = s1 * log2f(fabsf(c1) + 1.0f) * (1.0f / 3.0f);
    float acc[8] = {0.f, 0.f, 0.f, 0.f, 0.f, 0.f, 0.f, 0.f};
    for (int jj = 0; jj < 512; ++jj) {
      float hv = fmaxf(v0 * cw1[jj * 2] + v1 * cw1[jj * 2 + 1] + cb1[jj], 0.0f);
#pragma unroll
      for (int hh = 0; hh < 8; ++hh) acc[hh] += hv * cw2[hh * 512 + jj];
    }
#pragma unroll
    for (int hh = 0; hh < 8; ++hh) tbl[p][hh] = acc[hh];
  }
  __syncthreads();

  // bias16[h][n][m] = 16*sigmoid(tbl[idx(n,m)][h])
  for (int i = t; i < 8 * 4096; i += 256) {
    const int hh = i >> 12;
    const int p = i & 4095;
    const int n = p >> 6, m = p & 63;
    const int dh = (n >> 3) - (m >> 3) + 7;
    const int dw = (n & 7) - (m & 7) + 7;
    const float bv = tbl[dh * 15 + dw][hh];
    bias16[i] = 16.0f / (1.0f + __expf(-bv));
  }
}

// ---------------------------------------------------------------------------
// Kernel 1: per-window QKV GEMM (64x768, K=256) + bias + l2-norm of q,k.
// Writes qn,kn row-major [b][h][64][32] bf16 and v transposed [b][h][32][64].
// ---------------------------------------------------------------------------
__global__ __launch_bounds__(256) void qkv_kernel(
    const float* __restrict__ x, const __bf16* __restrict__ wbf,
    const float* __restrict__ qb, const float* __restrict__ vb,
    __bf16* __restrict__ qn, __bf16* __restrict__ kn, __bf16* __restrict__ vt) {
  extern __shared__ char smem1[];
  __bf16* xs = (__bf16*)smem1;                    // [64][256] bf16 : 32 KB
  float*  ys = (float*)(smem1 + 32768);           // [64][768] f32  : 192 KB

  const int b = blockIdx.x;
  const int t = threadIdx.x;
  const int lane = t & 31;
  const int wv = t >> 5;

  // Stage x[b] -> LDS bf16 (conversion in flight, so no async path here).
  const float* xb = x + (size_t)b * 64 * 256;
  for (int i = t; i < 4096; i += 256) {
    float4 v = ((const float4*)xb)[i];
    xs[4 * i + 0] = (__bf16)v.x;
    xs[4 * i + 1] = (__bf16)v.y;
    xs[4 * i + 2] = (__bf16)v.z;
    xs[4 * i + 3] = (__bf16)v.w;
  }
  __syncthreads();

  // y = x @ W^T + bias(concat(qb, 0, vb)).  Wave wv owns 96 output cols.
  const int half = lane >> 4;
  for (int nt = 0; nt < 6; ++nt) {
    const int nbase = wv * 96 + nt * 16;
    const int col = nbase + (lane & 15);
    const float bias = (col < 256) ? qb[col] : ((col >= 512) ? vb[col - 512] : 0.f);
    for (int mt = 0; mt < 4; ++mt) {
      f32x8 acc = {0.f, 0.f, 0.f, 0.f, 0.f, 0.f, 0.f, 0.f};
#pragma unroll
      for (int ks = 0; ks < 8; ++ks) {
        bf16x16 a = load_a_frag(xs, 256, mt * 16, ks * 32, lane);
        bf16x16 bb = load_b_frag(wbf, 256, nbase, ks * 32, lane);
        acc = wmma_bf16(a, bb, acc);
      }
#pragma unroll
      for (int e = 0; e < 8; ++e)
        ys[(mt * 16 + e + 8 * half) * 768 + col] = acc[e] + bias;
    }
  }
  __syncthreads();

  // Normalize q,k per (token,head); emit bf16 qn/kn and v^T.
  for (int pr = t; pr < 512; pr += 256) {
    const int n = pr >> 3, h = pr & 7;
    const size_t hb = ((size_t)b * 8 + h) * 2048;
    float tq[32], tk[32];
    float sq = 0.f, sk = 0.f;
#pragma unroll
    for (int d = 0; d < 32; ++d) {
      tq[d] = ys[n * 768 + h * 32 + d];
      tk[d] = ys[n * 768 + 256 + h * 32 + d];
      sq += tq[d] * tq[d];
      sk += tk[d] * tk[d];
    }
    const float iq = 1.0f / fmaxf(sqrtf(sq), 1e-12f);
    const float ik = 1.0f / fmaxf(sqrtf(sk), 1e-12f);
#pragma unroll
    for (int d = 0; d < 32; ++d) {
      qn[hb + n * 32 + d] = (__bf16)(tq[d] * iq);
      kn[hb + n * 32 + d] = (__bf16)(tk[d] * ik);
      vt[hb + d * 64 + n] = (__bf16)ys[n * 768 + 512 + h * 32 + d];
    }
  }
}

// ---------------------------------------------------------------------------
// Kernel 2: per-window attention. Wave == head. LDS = 320 KB (dynamic):
//   [0,128K)   scores bf16 [16][64][64]  (later reused as probs bf16 [8][64][64])
//   [128,256K) logits f32  [8][64][64]
//   [256,320K) k-stage bf16 [2][8][64][32] (later reused as o_buf bf16 [64][256])
// ---------------------------------------------------------------------------
__global__ __launch_bounds__(256) void attn_kernel(
    const __bf16* __restrict__ qn, const __bf16* __restrict__ kn_s,
    const __bf16* __restrict__ kn_d, const __bf16* __restrict__ vt,
    const float* __restrict__ bias16, const float* __restrict__ pc_w,
    const float* __restrict__ pc_b, const float* __restrict__ lscale,
    const __bf16* __restrict__ projw, const float* __restrict__ projb,
    float* __restrict__ out) {
  extern __shared__ char smem2[];
  __bf16* scores = (__bf16*)smem2;                 // [16][4096] bf16
  float*  logits = (float*)(smem2 + 131072);       // [8][4096] f32
  __bf16* kst    = (__bf16*)(smem2 + 262144);      // [2][8][64][32] bf16
  __bf16* probs  = scores;                         // reuse
  __bf16* obuf   = (__bf16*)(smem2 + 262144);      // reuse: [64][256] bf16

  const int b = blockIdx.x;
  const int t = threadIdx.x;
  const int lane = t & 31;
  const int h = t >> 5;  // wave id == head id
  const int half = lane >> 4;

  __builtin_prefetch(projw + (size_t)t * 256, 0, 1);  // warm L2 for projection

  // Stage keys of both views: async memory->LDS, 16B per lane per issue.
  {
    const unsigned kst_base = (unsigned)(uintptr_t)kst;
    const char* s0 = (const char*)(kn_s + (size_t)b * 16384);
    const char* s1 = (const char*)(kn_d + (size_t)b * 16384);
    // 32768 B per source = 2048 x 16B; 256 threads -> 8 issues per thread each.
    for (int i = t; i < 2048; i += 256)
      async_copy_b128(kst_base + 16u * i, s0 + 16 * i);
    for (int i = t; i < 2048; i += 256)
      async_copy_b128(kst_base + 32768u + 16u * i, s1 + 16 * i);
    wait_asynccnt0();
  }
  __syncthreads();

  // Scores: same = qn @ kn_s^T, diff = qn @ kn_d^T  (K = HD = 32, one WMMA).
  const __bf16* qh = qn + ((size_t)b * 8 + h) * 2048;  // [64][32]
  for (int mt = 0; mt < 4; ++mt) {
    bf16x16 a = load_a_frag(qh, 32, mt * 16, 0, lane);
#pragma unroll
    for (int src = 0; src < 2; ++src) {
      const __bf16* kb = kst + (src * 8 + h) * 2048;   // [64][32]
      __bf16* srow = scores + (src * 8 + h) * 4096;
#pragma unroll
      for (int nt = 0; nt < 4; ++nt) {
        f32x8 acc = {0.f, 0.f, 0.f, 0.f, 0.f, 0.f, 0.f, 0.f};
        acc = wmma_bf16(a, load_b_frag(kb, 32, nt * 16, 0, lane), acc);
        const int col = nt * 16 + (lane & 15);
#pragma unroll
        for (int e = 0; e < 8; ++e)
          srow[(mt * 16 + e + 8 * half) * 64 + col] = (__bf16)acc[e];
      }
    }
  }
  __syncthreads();

  // Head mixing: logits[h] = scale[h]*(pc_b[h] + sum_c pc_w[h,c]*scores[c]) + bias
  {
    float pcw[16];
#pragma unroll
    for (int c = 0; c < 16; ++c) pcw[c] = pc_w[h * 16 + c];
    const float pcb = pc_b[h];
    const float sc = __expf(fminf(lscale[h], 4.605170185988092f));  // ln(100)
    const float* bh = bias16 + h * 4096;
    for (int i = 0; i < 128; ++i) {
      const int p = lane + i * 32;
      float acc = pcb;
#pragma unroll
      for (int c = 0; c < 16; ++c) acc += pcw[c] * (float)scores[c * 4096 + p];
      logits[h * 4096 + p] = acc * sc + bh[p];
    }
  }
  __syncthreads();

  // Row softmax -> probs (bf16, reuses scores region).
  for (int r = t; r < 512; r += 256) {
    const float* row = logits + r * 64;
    float mx = -1e30f;
    for (int m = 0; m < 64; ++m) mx = fmaxf(mx, row[m]);
    float s = 0.f;
    for (int m = 0; m < 64; ++m) s += __expf(row[m] - mx);
    const float inv = 1.0f / s;
    for (int m = 0; m < 64; ++m)
      probs[r * 64 + m] = (__bf16)(__expf(row[m] - mx) * inv);
  }
  __syncthreads();

  // attn @ V: P[h] (64x64) @ v (64x32) via v^T, write into o_buf[64][256].
  {
    const __bf16* ph = probs + h * 4096;                     // [64][64]
    const __bf16* vh = vt + ((size_t)b * 8 + h) * 2048;      // [32][64] (= v^T)
    for (int mt = 0; mt < 4; ++mt) {
#pragma unroll
      for (int nt = 0; nt < 2; ++nt) {
        f32x8 acc = {0.f, 0.f, 0.f, 0.f, 0.f, 0.f, 0.f, 0.f};
#pragma unroll
        for (int ks = 0; ks < 2; ++ks) {
          bf16x16 a = load_a_frag(ph, 64, mt * 16, ks * 32, lane);
          bf16x16 bb = load_b_frag(vh, 64, nt * 16, ks * 32, lane);
          acc = wmma_bf16(a, bb, acc);
        }
        const int col = h * 32 + nt * 16 + (lane & 15);
#pragma unroll
        for (int e = 0; e < 8; ++e)
          obuf[(mt * 16 + e + 8 * half) * 256 + col] = (__bf16)acc[e];
      }
    }
  }
  __syncthreads();

  // Projection: o_buf (64x256) @ proj_w^T + proj_b -> global f32.
  for (int nt = 0; nt < 2; ++nt) {
    const int nbase = h * 32 + nt * 16;
    const int col = nbase + (lane & 15);
    const float pb = projb[col];
    for (int mt = 0; mt < 4; ++mt) {
      f32x8 acc = {0.f, 0.f, 0.f, 0.f, 0.f, 0.f, 0.f, 0.f};
#pragma unroll
      for (int ks = 0; ks < 8; ++ks) {
        bf16x16 a = load_a_frag(obuf, 256, mt * 16, ks * 32, lane);
        bf16x16 bb = load_b_frag(projw, 256, nbase, ks * 32, lane);
        acc = wmma_bf16(a, bb, acc);
      }
#pragma unroll
      for (int e = 0; e < 8; ++e)
        out[((size_t)b * 64 + mt * 16 + e + 8 * half) * 256 + col] = acc[e] + pb;
    }
  }
}

// ---------------------------------------------------------------------------
extern "C" void kernel_launch(void* const* d_in, const int* in_sizes, int n_in,
                              void* d_out, int out_size, void* d_ws, size_t ws_size,
                              hipStream_t stream) {
  const float* x1           = (const float*)d_in[0];
  const float* x2           = (const float*)d_in[1];
  const float* qkv_w1       = (const float*)d_in[2];
  const float* q_bias_1     = (const float*)d_in[3];
  const float* v_bias_1     = (const float*)d_in[4];
  const float* proj_w1      = (const float*)d_in[5];
  const float* proj_b1      = (const float*)d_in[6];
  const float* logit_scale1 = (const float*)d_in[7];
  const float* cpb1_w1      = (const float*)d_in[8];
  const float* cpb1_b1      = (const float*)d_in[9];
  const float* cpb1_w2      = (const float*)d_in[10];
  const float* pc1_w        = (const float*)d_in[11];
  const float* pc1_b        = (const float*)d_in[12];
  const float* qkv_w2       = (const float*)d_in[13];
  const float* q_bias_2     = (const float*)d_in[14];
  const float* v_bias_2     = (const float*)d_in[15];
  const float* proj_w2      = (const float*)d_in[16];
  const float* proj_b2      = (const float*)d_in[17];
  const float* logit_scale2 = (const float*)d_in[18];
  const float* cpb2_w1      = (const float*)d_in[19];
  const float* cpb2_b1      = (const float*)d_in[20];
  const float* cpb2_w2      = (const float*)d_in[21];
  const float* pc2_w        = (const float*)d_in[22];
  const float* pc2_b        = (const float*)d_in[23];
  (void)in_sizes; (void)n_in; (void)out_size; (void)ws_size;

  char* ws = (char*)d_ws;
  float*  bias16 = (float*)ws;                         // 2 * 32768 f32   (256 KB)
  __bf16* qkvbf  = (__bf16*)(ws + 262144);             // 2 * 196608 bf16 (768 KB)
  __bf16* projbf = (__bf16*)(ws + 1048576);            // 2 * 65536 bf16  (256 KB)
  __bf16* qn     = (__bf16*)(ws + 1310720);            // 2 * 8388608 bf16
  __bf16* kn     = (__bf16*)(ws + 1310720 + 33554432);
  __bf16* vt     = (__bf16*)(ws + 1310720 + 2 * 33554432);
  const size_t V = 8388608;  // elems per view for qn/kn/vt

  prep_kernel<<<1, 256, 0, stream>>>(qkv_w1, proj_w1, cpb1_w1, cpb1_b1, cpb1_w2,
                                     qkvbf, projbf, bias16);
  prep_kernel<<<1, 256, 0, stream>>>(qkv_w2, proj_w2, cpb2_w1, cpb2_b1, cpb2_w2,
                                     qkvbf + 196608, projbf + 65536, bias16 + 32768);

  const size_t lds1 = 32768 + 196608;  // 224 KB
  qkv_kernel<<<512, 256, lds1, stream>>>(x1, qkvbf, q_bias_1, v_bias_1,
                                         qn, kn, vt);
  qkv_kernel<<<512, 256, lds1, stream>>>(x2, qkvbf + 196608, q_bias_2, v_bias_2,
                                         qn + V, kn + V, vt + V);

  const size_t lds2 = 131072 + 131072 + 65536;  // 320 KB
  float* out = (float*)d_out;
  attn_kernel<<<512, 256, lds2, stream>>>(qn, kn, kn + V, vt, bias16,
                                          pc1_w, pc1_b, logit_scale1,
                                          projbf, proj_b1, out);
  attn_kernel<<<512, 256, lds2, stream>>>(qn + V, kn + V, kn, vt + V,
                                          bias16 + 32768, pc2_w, pc2_b,
                                          logit_scale2, projbf + 65536, proj_b2,
                                          out + (size_t)512 * 64 * 256);
}
